// DLPTLayer_9612136808567
// MI455X (gfx1250) — matrix operand order repaired and checked by reading, not compiled, not run on hardware
//
#include <hip/hip_runtime.h>
#include <hip/hip_bf16.h>
#include <cstdint>
#include <cstddef>

typedef _Float16 f16;
typedef __attribute__((ext_vector_type(16))) _Float16 v16h;
typedef __attribute__((ext_vector_type(8)))  float    v8f;

static constexpr size_t alignup(size_t x) { return (x + 31) & ~(size_t)31; }

// B-fragment swizzle slot for consumer GEMM with K-dim Kc (mult of 32):
// logical B[k][n] lives at ((ntile*(Kc/32)+kchunk)*32 + lane)*16 + (k&15),
// lane = ((k>>4)&1)*16 | (n&15).  Each lane's 16 values are contiguous (32B).
__device__ __forceinline__ int bswz(int k, int n, int Kc) {
  return (((n >> 4) * (Kc >> 5) + (k >> 5)) * 32 + (((k >> 4) & 1) << 4) + (n & 15)) * 16 + (k & 15);
}

// wave32 sum over all lanes (inactive contributors must pass 0)
__device__ __forceinline__ float wsum32(float v) {
  #pragma unroll
  for (int m = 16; m; m >>= 1) v += __shfl_xor(v, m, 32);
  return v;
}

// ---------------------------------------------------------------------------
// LDS layout (one wave == one cluster). Weight matrices stored B-swizzled f16.
// ---------------------------------------------------------------------------
template<int S, int DPE, int DF, int DE, int KX, int SP>
struct Lay {
  static constexpr int KSRC = DPE + DF;
  static constexpr size_t oW1B = 0;
  static constexpr size_t oW2B = alignup(oW1B + sizeof(f16) * KX * DE);
  static constexpr size_t oWQ  = alignup(oW2B + sizeof(f16) * KX * DE);
  static constexpr size_t oWK  = alignup(oWQ  + sizeof(f16) * DE * DE);
  static constexpr size_t oWV  = alignup(oWK  + sizeof(f16) * DE * DE);
  static constexpr size_t oWO  = alignup(oWV  + sizeof(f16) * DE * DE);
  static constexpr size_t oW1A = alignup(oWO  + sizeof(f16) * DE * DE);
  static constexpr size_t oB1A = alignup(oW1A + sizeof(float) * 4 * DPE);
  static constexpr size_t oW2A = alignup(oB1A + sizeof(float) * DPE);
  static constexpr size_t oB2A = alignup(oW2A + sizeof(float) * 6 * DPE);
  static constexpr size_t oB1B = alignup(oB2A + sizeof(float) * DPE);
  static constexpr size_t oB2B = alignup(oB1B + sizeof(float) * DE);
  static constexpr size_t oBQ  = alignup(oB2B + sizeof(float) * DE);
  static constexpr size_t oBK  = alignup(oBQ  + sizeof(float) * DE);
  static constexpr size_t oBV  = alignup(oBK  + sizeof(float) * DE);
  static constexpr size_t oBO  = alignup(oBV  + sizeof(float) * DE);
  static constexpr size_t oLNG = alignup(oBO  + sizeof(float) * DE);
  static constexpr size_t oLNB = alignup(oLNG + sizeof(float) * DE);
  static constexpr size_t oXB  = alignup(oLNB + sizeof(float) * DE);
  static constexpr size_t oHPO = alignup(oXB  + sizeof(f16)   * S * KX);
  static constexpr size_t oHGE = alignup(oHPO + sizeof(float) * S * DE);
  static constexpr size_t oQB  = alignup(oHGE + sizeof(f16)   * S * DE);
  static constexpr size_t oKB  = alignup(oQB  + sizeof(f16)   * S * DE);
  static constexpr size_t oHPH = alignup(oKB  + sizeof(f16)   * S * DE);
  static constexpr size_t oVB  = alignup(oHPH + sizeof(f16)   * S * DE);   // swizzled, Kc=SP
  static constexpr size_t oSC  = alignup(oVB  + sizeof(f16)   * SP * DE);
  static constexpr size_t oATT = alignup(oSC  + sizeof(float) * S * S);
  static constexpr size_t oO1  = alignup(oATT + sizeof(f16)   * S * SP);
  static constexpr size_t oOB  = alignup(oO1  + sizeof(f16)   * S * DE);
  static constexpr size_t total= alignup(oOB  + sizeof(float) * S * DE);
};

enum BKind { B_TRANS, B_SWZ };

// A fragment: rows m0..m0+15, K-chunk k0 (A row-major, lda == K).
// contiguous K order (k = 16*(lane>>4) + i) -> one contiguous 32B run per lane.
template<int K>
__device__ __forceinline__ v16h load_a_frag(const f16* __restrict__ A, int m0, int k0, int lane) {
  const f16* pa = A + (m0 + (lane & 15)) * K + k0 + (lane >> 4) * 16;
  v16h a;
  #pragma unroll
  for (int i = 0; i < 16; ++i) a[i] = pa[i];
  return a;
}

template<int N, int K, BKind BK>
__device__ __forceinline__ v16h load_b_frag(const f16* __restrict__ B, int n0, int k0, int lane) {
  if (BK == B_SWZ) {
    const f16* pb = B + (size_t)(((n0 >> 4) * (K >> 5) + (k0 >> 5)) * 32 + lane) * 16;
    return *(const v16h*)pb;                   // one aligned 32B LDS vector load
  } else {                                     // B given as NxK row-major (q @ k^T)
    const f16* pb = B + (n0 + (lane & 15)) * K + k0 + (lane >> 4) * 16;
    v16h b;
    #pragma unroll
    for (int i = 0; i < 16; ++i) b[i] = pb[i]; // contiguous along K
    return b;
  }
}

//   C[M x N](f32, row-major) = A[M x K] * B (+bias) * scale
template<int M, int N, int K, BKind BK, bool HB>
__device__ __forceinline__ void wmma_gemm_f32(const f16* __restrict__ A, const f16* __restrict__ B,
                                              float* __restrict__ C, const float* __restrict__ bias,
                                              float scale, int lane) {
  const int col = lane & 15, hid = lane >> 4;
  #pragma unroll
  for (int n0 = 0; n0 < N; n0 += 16) {
    const float bb = HB ? bias[n0 + col] : 0.0f;
    #pragma unroll
    for (int m0 = 0; m0 < M; m0 += 16) {
      v8f acc = {};
      #pragma unroll
      for (int k0 = 0; k0 < K; k0 += 32)
        acc = __builtin_amdgcn_wmma_f32_16x16x32_f16(false, load_a_frag<K>(A, m0, k0, lane),
                                                     false, load_b_frag<N, K, BK>(B, n0, k0, lane),
                                                     (short)0, acc, false, false);
      float* pc = C + (m0 + hid * 8) * N + (n0 + col);
      #pragma unroll
      for (int r = 0; r < 8; ++r) pc[r * N] = (acc[r] + bb) * scale;
    }
  }
}

// f16 output, row-major C
template<int M, int N, int K, BKind BK, bool HB>
__device__ __forceinline__ void wmma_gemm_f16(const f16* __restrict__ A, const f16* __restrict__ B,
                                              f16* __restrict__ C, const float* __restrict__ bias,
                                              float scale, int lane) {
  const int col = lane & 15, hid = lane >> 4;
  #pragma unroll
  for (int n0 = 0; n0 < N; n0 += 16) {
    const float bb = HB ? bias[n0 + col] : 0.0f;
    #pragma unroll
    for (int m0 = 0; m0 < M; m0 += 16) {
      v8f acc = {};
      #pragma unroll
      for (int k0 = 0; k0 < K; k0 += 32)
        acc = __builtin_amdgcn_wmma_f32_16x16x32_f16(false, load_a_frag<K>(A, m0, k0, lane),
                                                     false, load_b_frag<N, K, BK>(B, n0, k0, lane),
                                                     (short)0, acc, false, false);
      f16* pc = C + (m0 + hid * 8) * N + (n0 + col);
      #pragma unroll
      for (int r = 0; r < 8; ++r) pc[r * N] = (f16)((acc[r] + bb) * scale);
    }
  }
}

// dual output: f32 C (residual path) + f16 Ch (next GEMM's A operand)
template<int M, int N, int K, BKind BK>
__device__ __forceinline__ void wmma_gemm_dual(const f16* __restrict__ A, const f16* __restrict__ B,
                                               float* __restrict__ C, f16* __restrict__ Ch,
                                               const float* __restrict__ bias, int lane) {
  const int col = lane & 15, hid = lane >> 4;
  #pragma unroll
  for (int n0 = 0; n0 < N; n0 += 16) {
    const float bb = bias[n0 + col];
    #pragma unroll
    for (int m0 = 0; m0 < M; m0 += 16) {
      v8f acc = {};
      #pragma unroll
      for (int k0 = 0; k0 < K; k0 += 32)
        acc = __builtin_amdgcn_wmma_f32_16x16x32_f16(false, load_a_frag<K>(A, m0, k0, lane),
                                                     false, load_b_frag<N, K, BK>(B, n0, k0, lane),
                                                     (short)0, acc, false, false);
      float* pc = C  + (m0 + hid * 8) * N + (n0 + col);
      f16*   ph = Ch + (m0 + hid * 8) * N + (n0 + col);
      #pragma unroll
      for (int r = 0; r < 8; ++r) {
        const float v = acc[r] + bb;
        pc[r * N] = v;
        ph[r * N] = (f16)v;
      }
    }
  }
}

// fused two-output GEMM sharing the A operand (Q and K projections)
template<int M, int N, int K, BKind BK>
__device__ __forceinline__ void wmma_gemm2_f16(const f16* __restrict__ A,
                                               const f16* __restrict__ B1, const f16* __restrict__ B2,
                                               f16* __restrict__ C1, f16* __restrict__ C2,
                                               const float* __restrict__ bias1,
                                               const float* __restrict__ bias2,
                                               float scale1, int lane) {
  const int col = lane & 15, hid = lane >> 4;
  #pragma unroll
  for (int n0 = 0; n0 < N; n0 += 16) {
    const float bb1 = bias1[n0 + col], bb2 = bias2[n0 + col];
    #pragma unroll
    for (int m0 = 0; m0 < M; m0 += 16) {
      v8f a1 = {}, a2 = {};
      #pragma unroll
      for (int k0 = 0; k0 < K; k0 += 32) {
        const v16h av = load_a_frag<K>(A, m0, k0, lane);
        a1 = __builtin_amdgcn_wmma_f32_16x16x32_f16(false, av, false,
                 load_b_frag<N, K, BK>(B1, n0, k0, lane), (short)0, a1, false, false);
        a2 = __builtin_amdgcn_wmma_f32_16x16x32_f16(false, av, false,
                 load_b_frag<N, K, BK>(B2, n0, k0, lane), (short)0, a2, false, false);
      }
      f16* p1 = C1 + (m0 + hid * 8) * N + (n0 + col);
      f16* p2 = C2 + (m0 + hid * 8) * N + (n0 + col);
      #pragma unroll
      for (int r = 0; r < 8; ++r) {
        p1[r * N] = (f16)((a1[r] + bb1) * scale1);
        p2[r * N] = (f16)(a2[r] + bb2);
      }
    }
  }
}

// f16 output stored B-swizzled for a consumer GEMM whose K-dim is CK (V matrix)
template<int M, int N, int K, BKind BK, int CK>
__device__ __forceinline__ void wmma_gemm_f16_swzout(const f16* __restrict__ A, const f16* __restrict__ B,
                                                     f16* __restrict__ C, const float* __restrict__ bias,
                                                     int lane) {
  const int col = lane & 15, hid = lane >> 4;
  #pragma unroll
  for (int n0 = 0; n0 < N; n0 += 16) {
    const float bb = bias[n0 + col];
    #pragma unroll
    for (int m0 = 0; m0 < M; m0 += 16) {
      v8f acc = {};
      #pragma unroll
      for (int k0 = 0; k0 < K; k0 += 32)
        acc = __builtin_amdgcn_wmma_f32_16x16x32_f16(false, load_a_frag<K>(A, m0, k0, lane),
                                                     false, load_b_frag<N, K, BK>(B, n0, k0, lane),
                                                     (short)0, acc, false, false);
      #pragma unroll
      for (int r = 0; r < 8; ++r) {
        const int m = m0 + hid * 8 + r, n = n0 + col;
        C[bswz(m, n, CK)] = (f16)(acc[r] + bb);
      }
    }
  }
}

struct BlockParams {
  const float *w1a, *b1a, *w1b, *b1b, *w2a, *b2a, *w2b, *b2b;
  const float *wq, *bq, *wk, *bk, *wv, *bv, *wo, *bo, *lng, *lnb;
};

static __device__ __forceinline__ void wave_copy_f32(float* dst, const float* src, int n, int lane) {
  for (int i = lane; i < n; i += 32) dst[i] = src[i];
}
// stage [ksrc x N] f32 weight -> [K x N] f16, zero K-padded, in B-fragment order.
template<int K, int N>
__device__ __forceinline__ void stage_w_swz(f16* dst, const float* src, int ksrc, int lane) {
  for (int idx = lane; idx < K * N; idx += 32) {
    const int k = idx / N, n = idx - k * N;
    dst[bswz(k, n, K)] = (k < ksrc) ? (f16)src[idx] : (f16)0;
  }
}

// ---------------------------------------------------------------------------
// One DLPT block. Grid = B*(N/S) workgroups of one wave. out[perm[j]] = h[j]
// (equivalent to the reference's gather through argsort(perm)).
// ---------------------------------------------------------------------------
template<int S, int DPE, int DF, int DE, int KX, int SP>
__global__ __launch_bounds__(32, 1)
void dlpt_block(const float* __restrict__ pos, const float* __restrict__ feat,
                const int* __restrict__ perm, BlockParams P,
                float* __restrict__ out, int Bn, int Nn) {
  using L = Lay<S, DPE, DF, DE, KX, SP>;
  extern __shared__ char smem[];
  f16*   W1B  = (f16*)  (smem + L::oW1B);
  f16*   W2B  = (f16*)  (smem + L::oW2B);
  f16*   WQ   = (f16*)  (smem + L::oWQ);
  f16*   WK   = (f16*)  (smem + L::oWK);
  f16*   WV   = (f16*)  (smem + L::oWV);
  f16*   WO   = (f16*)  (smem + L::oWO);
  float* SW1A = (float*)(smem + L::oW1A);
  float* SB1A = (float*)(smem + L::oB1A);
  float* SW2A = (float*)(smem + L::oW2A);
  float* SB2A = (float*)(smem + L::oB2A);
  float* SB1B = (float*)(smem + L::oB1B);
  float* SB2B = (float*)(smem + L::oB2B);
  float* SBQ  = (float*)(smem + L::oBQ);
  float* SBK  = (float*)(smem + L::oBK);
  float* SBV  = (float*)(smem + L::oBV);
  float* SBO  = (float*)(smem + L::oBO);
  float* SLNG = (float*)(smem + L::oLNG);
  float* SLNB = (float*)(smem + L::oLNB);
  f16*   XB   = (f16*)  (smem + L::oXB);
  float* HPO  = (float*)(smem + L::oHPO);
  f16*   HGE  = (f16*)  (smem + L::oHGE);
  f16*   QB   = (f16*)  (smem + L::oQB);
  f16*   KB   = (f16*)  (smem + L::oKB);
  f16*   HPH  = (f16*)  (smem + L::oHPH);
  f16*   VB   = (f16*)  (smem + L::oVB);   // B-swizzled (Kc = SP)
  float* SC   = (float*)(smem + L::oSC);
  f16*   ATT  = (f16*)  (smem + L::oATT);
  f16*   O1   = (f16*)  (smem + L::oO1);
  float* OB   = (float*)(smem + L::oOB);

  const int lane = threadIdx.x;

  // ---- stage weights (B-swizzled f16) and biases (f32) ----
  stage_w_swz<KX, DE>(W1B, P.w1b, L::KSRC, lane);
  stage_w_swz<KX, DE>(W2B, P.w2b, L::KSRC, lane);
  stage_w_swz<DE, DE>(WQ, P.wq, DE, lane);
  stage_w_swz<DE, DE>(WK, P.wk, DE, lane);
  stage_w_swz<DE, DE>(WV, P.wv, DE, lane);
  stage_w_swz<DE, DE>(WO, P.wo, DE, lane);
  wave_copy_f32(SW1A, P.w1a, 4 * DPE, lane);
  wave_copy_f32(SB1A, P.b1a, DPE, lane);
  wave_copy_f32(SW2A, P.w2a, 6 * DPE, lane);
  wave_copy_f32(SB2A, P.b2a, DPE, lane);
  wave_copy_f32(SB1B, P.b1b, DE, lane);
  wave_copy_f32(SB2B, P.b2b, DE, lane);
  wave_copy_f32(SBQ, P.bq, DE, lane);
  wave_copy_f32(SBK, P.bk, DE, lane);
  wave_copy_f32(SBV, P.bv, DE, lane);
  wave_copy_f32(SBO, P.bo, DE, lane);
  wave_copy_f32(SLNG, P.lng, DE, lane);
  wave_copy_f32(SLNB, P.lnb, DE, lane);
  if (SP > S) {  // zero swizzled V (pad K rows) with wide stores
    uint4* vz = (uint4*)VB;
    for (int i = lane; i < (int)(SP * DE * sizeof(f16) / sizeof(uint4)); i += 32)
      vz[i] = make_uint4(0, 0, 0, 0);
  }

  const int nc = Nn / S;
  const int b  = blockIdx.x / nc;
  const int c  = blockIdx.x % nc;
  const int s  = lane;
  const bool act = (s < S);

  // ---- gather cluster point (registers), features straight into X (f16) ----
  int p = 0;
  float px = 0.f, py = 0.f, pz = 0.f;
  if (act) {
    p = perm[(size_t)b * Nn + (size_t)c * S + s];
    const float* pp = pos + ((size_t)b * Nn + p) * 3;
    px = pp[0]; py = pp[1]; pz = pp[2];
    const float* pf = feat + ((size_t)b * Nn + p) * DF;
    #pragma unroll
    for (int j = 0; j < DF; ++j)        XB[s * KX + DPE + j] = (f16)pf[j];
    #pragma unroll
    for (int j = DPE + DF; j < KX; ++j) XB[s * KX + j] = (f16)0;
  }

  // ---- cog / local coords / norm / avg via wave butterfly reductions ----
  const float rs = 1.0f / S;
  const float cx = wsum32(px) * rs, cy = wsum32(py) * rs, cz = wsum32(pz) * rs;
  float lpx = 0.f, lpy = 0.f, lpz = 0.f, nr = 0.f;
  if (act) {
    lpx = px - cx; lpy = py - cy; lpz = pz - cz;
    nr  = sqrtf(lpx * lpx + lpy * lpy + lpz * lpz);
  }
  const float ax = wsum32(lpx) * rs, ay = wsum32(lpy) * rs, az = wsum32(lpz) * rs;

  // ---- X1 first DPE cols: r = [lp,nrm] @ w1a + b1a ----
  if (act) {
    #pragma unroll
    for (int d = 0; d < DPE; ++d) {
      const float r = SB1A[d] + lpx * SW1A[0*DPE+d] + lpy * SW1A[1*DPE+d]
                    + lpz * SW1A[2*DPE+d] + nr * SW1A[3*DPE+d];
      XB[s * KX + d] = (f16)r;
    }
  }
  __syncthreads();

  // ---- h_pos = X1 @ w1b + b1b : dual store f32 (residual) + f16 (V's A) ----
  wmma_gemm_dual<S, DE, KX, B_SWZ>(XB, W1B, HPO, HPH, SB1B, lane);
  __syncthreads();

  // ---- X2: overwrite first DPE cols with r_hat = [avg(lp), lp] @ w2a + b2a ----
  if (act) {
    #pragma unroll
    for (int d = 0; d < DPE; ++d) {
      const float rh = SB2A[d]
        + ax  * SW2A[0*DPE+d] + ay  * SW2A[1*DPE+d] + az  * SW2A[2*DPE+d]
        + lpx * SW2A[3*DPE+d] + lpy * SW2A[4*DPE+d] + lpz * SW2A[5*DPE+d];
      XB[s * KX + d] = (f16)rh;
    }
  }
  __syncthreads();

  // ---- h_geo (f16); then q,k fused (shared A) and v (swizzle-on-store) ----
  wmma_gemm_f16<S, DE, KX, B_SWZ, true>(XB, W2B, HGE, SB2B, 1.0f, lane);
  __syncthreads();
  wmma_gemm2_f16<S, DE, DE, B_SWZ>(HGE, WQ, WK, QB, KB, SBQ, SBK, rsqrtf((float)DE), lane);
  wmma_gemm_f16_swzout<S, DE, DE, B_SWZ, SP>(HPH, WV, VB, SBV, lane);
  __syncthreads();

  // ---- scores = q @ k^T (transposed path loads contiguous K) ----
  wmma_gemm_f32<S, S, DE, B_TRANS, false>(QB, KB, SC, nullptr, 1.0f, lane);
  __syncthreads();

  // ---- softmax row -> ATT (packed 2xf16 b32 stores, K-padded to SP) ----
  if (act) {
    float ev[S];
    float mx = -3.0e38f;
    #pragma unroll
    for (int t = 0; t < S; ++t) { ev[t] = SC[s * S + t]; mx = fmaxf(mx, ev[t]); }
    float sum = 0.f;
    #pragma unroll
    for (int t = 0; t < S; ++t) { ev[t] = __expf(ev[t] - mx); sum += ev[t]; }
    const float inv = 1.0f / sum;
    unsigned int* a32 = (unsigned int*)ATT;
    #pragma unroll
    for (int t = 0; t < SP; t += 2) {
      const f16 e0 = (t     < S) ? (f16)(ev[t < S ? t : 0] * inv)     : (f16)0;
      const f16 e1 = (t + 1 < S) ? (f16)(ev[t + 1 < S ? t + 1 : 0] * inv) : (f16)0;
      const unsigned int u = ((unsigned int)__builtin_bit_cast(unsigned short, e1) << 16)
                           |  (unsigned int)__builtin_bit_cast(unsigned short, e0);
      a32[(s * SP + t) >> 1] = u;
    }
  }
  __syncthreads();

  // ---- o = (ATT @ V) @ wo + bo ----
  wmma_gemm_f16<S, DE, SP, B_SWZ, false>(ATT, VB, O1, nullptr, 1.0f, lane);
  __syncthreads();
  wmma_gemm_f32<S, DE, DE, B_SWZ, true>(O1, WO, OB, SBO, 1.0f, lane);
  __syncthreads();

  // ---- residual + LayerNorm in registers, scatter through perm ----
  if (act) {
    float hreg[DE];
    float mu = 0.f;
    #pragma unroll
    for (int e = 0; e < DE; ++e) { hreg[e] = HPO[s * DE + e] + OB[s * DE + e]; mu += hreg[e]; }
    mu *= (1.0f / DE);
    float var = 0.f;
    #pragma unroll
    for (int e = 0; e < DE; ++e) { const float d = hreg[e] - mu; var += d * d; }
    var *= (1.0f / DE);
    const float inv = rsqrtf(var + 1e-5f);
    float* po = out + ((size_t)b * Nn + p) * DE;
    #pragma unroll
    for (int e = 0; e < DE; ++e)
      po[e] = (hreg[e] - mu) * inv * SLNG[e] + SLNB[e];
  }
}

// ---------------------------------------------------------------------------
// Final fps_idx gather: pos_ds (B,M,3) then feat_ds (B,M,64) into d_out.
// ---------------------------------------------------------------------------
__global__ void dlpt_gather(const float* __restrict__ pos, const float* __restrict__ f2,
                            const int* __restrict__ fps, float* __restrict__ opos,
                            float* __restrict__ ofeat, int Bn, int Nn, int Mn) {
  const int i = blockIdx.x * blockDim.x + threadIdx.x;
  if (i >= Bn * Mn) return;
  const int b = i / Mn;
  const int p = fps[i];
  const float* sp = pos + ((size_t)b * Nn + p) * 3;
  float* dp = opos + (size_t)i * 3;
  dp[0] = sp[0]; dp[1] = sp[1]; dp[2] = sp[2];
  const float4* sf = (const float4*)(f2 + ((size_t)b * Nn + p) * 64);
  float4* df = (float4*)(ofeat + (size_t)i * 64);
  #pragma unroll
  for (int e = 0; e < 16; ++e) df[e] = sf[e];
}

extern "C" void kernel_launch(void* const* d_in, const int* in_sizes, int n_in,
                              void* d_out, int out_size, void* d_ws, size_t ws_size,
                              hipStream_t stream) {
  (void)in_sizes; (void)n_in; (void)out_size; (void)ws_size;
  const int B = 4, N = 65536, M = N / 4;   // shapes fixed by setup_inputs()

  const float* pos  = (const float*)d_in[0];
  const float* feat = (const float*)d_in[1];
  const int* perm1  = (const int*)d_in[2];
  const int* perm2  = (const int*)d_in[3];
  const int* fps    = (const int*)d_in[4];

  BlockParams P1, P2;
  {
    const float** q = (const float**)&P1;
    for (int j = 0; j < 18; ++j) q[j] = (const float*)d_in[5 + j];
    const float** r = (const float**)&P2;
    for (int j = 0; j < 18; ++j) r[j] = (const float*)d_in[23 + j];
  }

  float* f1 = (float*)d_ws;                                               // B*N*32 f32
  float* f2 = (float*)((char*)d_ws + (size_t)B * N * 32 * sizeof(float)); // B*N*64 f32

  constexpr size_t lds1 = Lay<16, 3, 3, 32, 32, 32>::total;
  constexpr size_t lds2 = Lay<32, 32, 32, 64, 64, 32>::total;

  // Block 1: S=16, dpe=3, df=3, de=32; X K-padded 6 -> 32.
  dlpt_block<16, 3, 3, 32, 32, 32>
      <<<B * (N / 16), 32, lds1, stream>>>(pos, feat, perm1, P1, f1, B, N);
  // Block 2: S=32, dpe=32, df=32, de=64; X K = 64 exact.
  dlpt_block<32, 32, 32, 64, 64, 32>
      <<<B * (N / 32), 32, lds2, stream>>>(pos, f1, perm2, P2, f2, B, N);

  float* opos  = (float*)d_out;
  float* ofeat = opos + (size_t)B * M * 3;
  dlpt_gather<<<(B * M + 255) / 256, 256, 0, stream>>>(pos, f2, fps, opos, ofeat, B, N, M);
}
